// RefractionSheet_77790447665466
// MI455X (gfx1250) — matrix operand order repaired and checked by reading, not compiled, not run on hardware
//
#include <hip/hip_runtime.h>
#include <math.h>

// ---------------------------------------------------------------------------
// RefractionSheet on MI455X (gfx1250): bf16 WMMA GEMM pipeline
//   - weights bf16-converted + K-padded in workspace
//   - GEMM: block stages one 16-col weight strip (full K) in LDS,
//     each wave computes 4 M-tiles (64x16) reusing the LDS B operand.
//     Operand loads batched ahead of the 4 WMMAs per K-step so the
//     scheduler can use partial s_wait_loadcnt and overlap VMEM with WMMA.
// ---------------------------------------------------------------------------

#define SIREN_CLAMP 25.132741228718345f
#define DEG2 0.03490658503988659f
#define TABLE_MASK 0x7FFFFu   // TABLE = 2^19
#define HASH_P1 2654435761u

typedef __attribute__((ext_vector_type(16))) __bf16 v16bf;
typedef __attribute__((ext_vector_type(8)))  float  v8f;

union ABVec { v16bf v; uint4 q[2]; };

__device__ __forceinline__ unsigned short f2bf(float f) {
    unsigned u = __float_as_uint(f);
    unsigned rb = 0x7FFFu + ((u >> 16) & 1u);        // round-to-nearest-even
    return (unsigned short)((u + rb) >> 16);
}

#define ACT_NONE    0
#define ACT_SIREN   1
#define ACT_RELU    2
#define ACT_SIGMOID 3

// ---------------------------------------------------------------------------
// GEMM: out[M,N] = act(A[M,K](bf16) @ W[N,K](bf16)^T + bias[N])
//   grid  = (M/512, ceil(N/16)), block = 256 (8 waves)
//   Block stages W strip rows [tn*16, tn*16+16) x K into LDS (row pad +8 elems
//   -> conflict-free ds_load_b128). Each wave computes 4 consecutive 16x16
//   M-tiles sharing one LDS B operand per K-step (v_wmma_f32_16x16x32_bf16).
//   K, lda, ldw multiples of 32; M multiple of 512; ragged N guarded on store.
// ---------------------------------------------------------------------------
__global__ void gemm_wmma_bf16(const unsigned short* __restrict__ A, int lda,
                               const unsigned short* __restrict__ Wt, int ldw,
                               const float* __restrict__ bias,
                               unsigned short* __restrict__ outB,
                               float* __restrict__ outF,
                               int ldo, int N, int K, int act)
{
    extern __shared__ unsigned short sW[];           // 16 x (K+8) bf16
    const int LDSK = K + 8;
    const int tn   = blockIdx.y;

    // ---- cooperative stage of the 16-row weight strip (clamped rows) ----
    const int chunksPerRow = K >> 3;                 // uint4 chunks per row
    for (int t = threadIdx.x; t < 16 * chunksPerRow; t += blockDim.x) {
        int row  = t / chunksPerRow;
        int kc   = t - row * chunksPerRow;
        int nRow = tn * 16 + row;
        if (nRow > N - 1) nRow = N - 1;
        *(uint4*)(sW + row * LDSK + kc * 8) =
            *(const uint4*)(Wt + (size_t)nRow * ldw + kc * 8);
    }
    __syncthreads();

    const int wave = threadIdx.x >> 5;
    const int lane = threadIdx.x & 31;
    const int r    = lane & 15;
    const int half = lane >> 4;

    // 4 M-tiles per wave: rows [tmBase*16, tmBase*16+64)
    const int tmBase = blockIdx.x * 32 + wave * 4;
    // A layout (16x32 bf16): per lane two 16B chunks at kb=half*8 and kb+16
    const unsigned short* aRow = A + ((size_t)tmBase * 16 + r) * lda + half * 8;
    // B layout (32x16 bf16): lane = column r, two 16B chunks at kb=half*8 and kb+16
    const unsigned short* sB   = sW + r * LDSK + half * 8;

    v8f acc[4];
    #pragma unroll
    for (int t = 0; t < 4; ++t)
        acc[t] = (v8f){0.f,0.f,0.f,0.f,0.f,0.f,0.f,0.f};

    for (int k0 = 0; k0 < K; k0 += 32) {
        // batch ALL operand loads first so WMMAs can overlap outstanding VMEM
        ABVec b, a[4];
        b.q[0] = *(const uint4*)(sB + k0);
        b.q[1] = *(const uint4*)(sB + k0 + 16);
        #pragma unroll
        for (int t = 0; t < 4; ++t) {
            const unsigned short* ap = aRow + (size_t)t * 16 * lda + k0;
            a[t].q[0] = *(const uint4*)(ap);
            a[t].q[1] = *(const uint4*)(ap + 16);
        }
        #pragma unroll
        for (int t = 0; t < 4; ++t)
            acc[t] = __builtin_amdgcn_wmma_f32_16x16x32_bf16(
                         false, a[t].v, false, b.v, (short)0, acc[t], false, false);
    }

    const int nGlob = tn * 16 + r;
    if (nGlob < N) {
        const float bv = bias[nGlob];
        #pragma unroll
        for (int t = 0; t < 4; ++t) {
            const int mBase = (tmBase + t) * 16 + half * 8;  // D: VGPR i -> row mBase+i
            #pragma unroll
            for (int i = 0; i < 8; ++i) {
                float v = acc[t][i] + bv;
                if (act == ACT_SIREN)        v = __sinf(fminf(fmaxf(v, -SIREN_CLAMP), SIREN_CLAMP));
                else if (act == ACT_RELU)    v = fmaxf(v, 0.f);
                else if (act == ACT_SIGMOID) v = 1.f / (1.f + __expf(-v));
                const size_t o = (size_t)(mBase + i) * ldo + nGlob;
                if (outF) outF[o] = v;
                else      outB[o] = f2bf(v);
            }
        }
    }
}

// ---------------------------------------------------------------------------
// Weight conversion: fp32 (N,K) -> bf16 (N,Kpad) zero-padded
// ---------------------------------------------------------------------------
__global__ void cvt_w_kernel(const float* __restrict__ src,
                             unsigned short* __restrict__ dst,
                             int N, int K, int Kpad)
{
    int idx = blockIdx.x * blockDim.x + threadIdx.x;
    if (idx >= N * Kpad) return;
    int n = idx / Kpad, k = idx % Kpad;
    dst[idx] = (k < K) ? f2bf(src[(size_t)n * K + k]) : (unsigned short)0;
}

// ---------------------------------------------------------------------------
// Hashgrid encode (16 levels x 2 feats) + camera embed -> bf16 (B,64)
// ---------------------------------------------------------------------------
__global__ void hash_input_kernel(const float* __restrict__ y,
                                  const float* __restrict__ x,
                                  const float* __restrict__ cam,
                                  const float* __restrict__ table,
                                  unsigned short* __restrict__ out,
                                  int Bn)
{
    int b = blockIdx.x * blockDim.x + threadIdx.x;
    if (b >= Bn) return;
    float yL = fminf(fmaxf(roundf(y[b]), 0.f), 511.f);
    float xL = fminf(fmaxf(roundf(x[b]), 0.f), 511.f);
    float u = xL * (1.f / 512.f);
    float v = yL * (1.f / 512.f);

    float s = 32.f;                     // BASE_RES * SCALE^l
    unsigned short* o = out + (size_t)b * 64;
    for (int l = 0; l < 16; ++l) {
        float res = floorf(s);
        float px = u * res, py = v * res;
        float fx = floorf(px), fy = floorf(py);
        float wx = px - fx,   wy = py - fy;
        unsigned cx = (unsigned)fx, cy = (unsigned)fy;
        unsigned i00 = (cx       ^ (cy       * HASH_P1)) & TABLE_MASK;
        unsigned i10 = ((cx + 1) ^ (cy       * HASH_P1)) & TABLE_MASK;
        unsigned i01 = (cx       ^ ((cy + 1) * HASH_P1)) & TABLE_MASK;
        unsigned i11 = ((cx + 1) ^ ((cy + 1) * HASH_P1)) & TABLE_MASK;
        const float* tl = table + (size_t)l * (TABLE_MASK + 1u) * 2u;
        #pragma unroll
        for (int f = 0; f < 2; ++f) {
            float f00 = tl[(size_t)i00 * 2 + f];
            float f10 = tl[(size_t)i10 * 2 + f];
            float f01 = tl[(size_t)i01 * 2 + f];
            float f11 = tl[(size_t)i11 * 2 + f];
            float val = f00 * (1.f - wx) * (1.f - wy) + f10 * wx * (1.f - wy)
                      + f01 * (1.f - wx) * wy         + f11 * wx * wy;
            o[l * 2 + f] = f2bf(val);
        }
        s *= 1.3819f;
    }
    for (int i = 0; i < 32; ++i) o[32 + i] = f2bf(cam[i]);
}

// ---------------------------------------------------------------------------
// positional encoding of a 3-vector: [v, sin(2^i v), cos(2^i v)] -> 51 values
// ---------------------------------------------------------------------------
__device__ __forceinline__ void enc3(const float* v, float* o)
{
    o[0] = v[0]; o[1] = v[1]; o[2] = v[2];
    float k = 1.f;
    #pragma unroll
    for (int i = 0; i < 8; ++i) {
        #pragma unroll
        for (int c = 0; c < 3; ++c) {
            o[3 + i * 6 + c]     = __sinf(k * v[c]);
            o[3 + i * 6 + 3 + c] = __cosf(k * v[c]);
        }
        k *= 2.f;
    }
}

// ---------------------------------------------------------------------------
// delta -> dirs -> PE; assemble head/alpha inputs (bf16, K padded 486->512)
// also writes delta_raw into d_out[4B .. 13B)
// ---------------------------------------------------------------------------
__global__ void assemble_kernel(const float* __restrict__ delta_lin, // (B,16), 9 valid
                                const float* __restrict__ ray_o,
                                const float* __restrict__ ray_d,
                                const unsigned short* __restrict__ base, // (B,384)
                                unsigned short* __restrict__ feat0,
                                unsigned short* __restrict__ feat1,
                                unsigned short* __restrict__ feat2,
                                float* __restrict__ dout, int Bn)
{
    int b = blockIdx.x * blockDim.x + threadIdx.x;
    if (b >= Bn) return;

    float dr[9];
    #pragma unroll
    for (int j = 0; j < 9; ++j) {
        float t = tanhf(delta_lin[(size_t)b * 16 + j]);
        t = fminf(fmaxf(t, -1.f), 1.f);
        dr[j] = t;
        dout[(size_t)4 * Bn + (size_t)b * 9 + j] = t;   // delta_raw output
    }
    float ro[3], rd[3];
    #pragma unroll
    for (int c = 0; c < 3; ++c) { ro[c] = ray_o[b * 3 + c]; rd[c] = ray_d[b * 3 + c]; }

    float peo[51];
    enc3(ro, peo);

    unsigned short* feats[3] = { feat0 + (size_t)b * 512,
                                 feat1 + (size_t)b * 512,
                                 feat2 + (size_t)b * 512 };
    const unsigned short* bb = base + (size_t)b * 384;

    #pragma unroll
    for (int d = 0; d < 3; ++d) {
        float v3[3];
        #pragma unroll
        for (int c = 0; c < 3; ++c) v3[c] = rd[c] + dr[d * 3 + c] * DEG2;
        float nrm = sqrtf(v3[0]*v3[0] + v3[1]*v3[1] + v3[2]*v3[2]);
        nrm = fmaxf(nrm, 1e-12f);
        float inv = 1.f / nrm;
        v3[0] *= inv; v3[1] *= inv; v3[2] *= inv;

        float ped[51];
        enc3(v3, ped);

        unsigned short* f = feats[d];
        for (int t = 0; t < 384; ++t) f[t]       = bb[t];
        for (int t = 0; t < 51;  ++t) f[384 + t] = f2bf(ped[t]);
        for (int t = 0; t < 51;  ++t) f[435 + t] = f2bf(peo[t]);
        for (int t = 0; t < 26;  ++t) f[486 + t] = 0;       // zero pad to 512
    }
}

// ---------------------------------------------------------------------------
// rgbR[b,i] = sum_c mixW[i,c] * head_c[b,i] (sigmoid already fused in GEMM)
// alpha[b]  = alpha_out[b,0]
// ---------------------------------------------------------------------------
__global__ void finalize_kernel(const float* __restrict__ h0,
                                const float* __restrict__ h1,
                                const float* __restrict__ h2,
                                const float* __restrict__ aout,
                                const float* __restrict__ mixW,
                                float* __restrict__ dout, int Bn)
{
    int b = blockIdx.x * blockDim.x + threadIdx.x;
    if (b >= Bn) return;
    #pragma unroll
    for (int i = 0; i < 3; ++i) {
        float v = mixW[i * 3 + 0] * h0[(size_t)b * 16 + i]
                + mixW[i * 3 + 1] * h1[(size_t)b * 16 + i]
                + mixW[i * 3 + 2] * h2[(size_t)b * 16 + i];
        dout[(size_t)b * 3 + i] = v;
    }
    dout[(size_t)3 * Bn + b] = aout[(size_t)b * 16];
}

// ---------------------------------------------------------------------------
// host: orchestrate pipeline
// ---------------------------------------------------------------------------
extern "C" void kernel_launch(void* const* d_in, const int* in_sizes, int n_in,
                              void* d_out, int out_size, void* d_ws, size_t ws_size,
                              hipStream_t stream)
{
    const int Bn = in_sizes[0];                 // 16384
    const float* y_in   = (const float*)d_in[0];
    const float* x_in   = (const float*)d_in[1];
    const float* ray_o  = (const float*)d_in[2];
    const float* ray_d  = (const float*)d_in[3];
    const float* cam    = (const float*)d_in[4];
    const float* htab   = (const float*)d_in[5];
    const float* mixW   = (const float*)d_in[52];
    float* dout = (float*)d_out;

    // ---- workspace carve (all offsets 256B aligned) ----
    char* ws = (char*)d_ws;
    size_t off = 0;
    auto alloc = [&](size_t bytes) -> void* {
        off = (off + 255) & ~(size_t)255;
        void* p = ws + off;
        off += bytes;
        return p;
    };
    typedef unsigned short u16;

    // bf16 weights (K padded to mult of 32)
    u16* posW0 = (u16*)alloc((size_t)512 * 64  * 2);
    u16* posW1 = (u16*)alloc((size_t)512 * 512 * 2);
    u16* posW2 = (u16*)alloc((size_t)384 * 512 * 2);
    u16* dW[5];
    dW[0] = (u16*)alloc((size_t)256 * 384 * 2);
    dW[1] = (u16*)alloc((size_t)256 * 256 * 2);
    dW[2] = (u16*)alloc((size_t)256 * 256 * 2);
    dW[3] = (u16*)alloc((size_t)256 * 256 * 2);
    dW[4] = (u16*)alloc((size_t)9   * 256 * 2);
    u16* hW[3][4];
    for (int c = 0; c < 3; ++c) {
        hW[c][0] = (u16*)alloc((size_t)768 * 512 * 2);   // 486 -> pad 512
        hW[c][1] = (u16*)alloc((size_t)768 * 768 * 2);
        hW[c][2] = (u16*)alloc((size_t)768 * 768 * 2);
        hW[c][3] = (u16*)alloc((size_t)3   * 768 * 2);
    }
    u16* aW0 = (u16*)alloc((size_t)256 * 512 * 2);       // 486 -> pad 512
    u16* aW1 = (u16*)alloc((size_t)256 * 256 * 2);
    u16* aW2 = (u16*)alloc((size_t)1   * 256 * 2);

    // bf16 activations
    u16* in0   = (u16*)alloc((size_t)Bn * 64  * 2);
    u16* bufA  = (u16*)alloc((size_t)Bn * 768 * 2);
    u16* bufB  = (u16*)alloc((size_t)Bn * 768 * 2);
    u16* base  = (u16*)alloc((size_t)Bn * 384 * 2);
    u16* feat0 = (u16*)alloc((size_t)Bn * 512 * 2);
    u16* feat1 = (u16*)alloc((size_t)Bn * 512 * 2);
    u16* feat2 = (u16*)alloc((size_t)Bn * 512 * 2);

    // f32 scratch
    float* delta_lin = (float*)alloc((size_t)Bn * 16 * 4);
    float* hout[3];
    hout[0] = (float*)alloc((size_t)Bn * 16 * 4);
    hout[1] = (float*)alloc((size_t)Bn * 16 * 4);
    hout[2] = (float*)alloc((size_t)Bn * 16 * 4);
    float* aout = (float*)alloc((size_t)Bn * 16 * 4);
    (void)ws_size; (void)n_in; (void)out_size;

    // ---- weight conversions ----
    auto cvt = [&](int idx, u16* dst, int N, int K, int Kpad) {
        int tot = N * Kpad;
        cvt_w_kernel<<<(tot + 255) / 256, 256, 0, stream>>>(
            (const float*)d_in[idx], dst, N, K, Kpad);
    };
    cvt(6,  posW0, 512, 64,  64);
    cvt(8,  posW1, 512, 512, 512);
    cvt(10, posW2, 384, 512, 512);
    cvt(12, dW[0], 256, 384, 384);
    cvt(14, dW[1], 256, 256, 256);
    cvt(16, dW[2], 256, 256, 256);
    cvt(18, dW[3], 256, 256, 256);
    cvt(20, dW[4], 9,   256, 256);
    for (int c = 0; c < 3; ++c) {
        int bi = 22 + 8 * c;
        cvt(bi + 0, hW[c][0], 768, 486, 512);
        cvt(bi + 2, hW[c][1], 768, 768, 768);
        cvt(bi + 4, hW[c][2], 768, 768, 768);
        cvt(bi + 6, hW[c][3], 3,   768, 768);
    }
    cvt(46, aW0, 256, 486, 512);
    cvt(48, aW1, 256, 256, 256);
    cvt(50, aW2, 1,   256, 256);

    // ---- GEMM launcher: grid (M/512, tilesN), LDS = 16*(K+8)*2 bytes ----
    auto gemm = [&](const u16* A, int lda, const u16* W, int ldw, int bias_idx,
                    u16* oB, float* oF, int ldo, int N, int K, int act) {
        dim3 grid(Bn >> 9, (N + 15) >> 4);
        size_t lds = (size_t)16 * (K + 8) * sizeof(unsigned short);
        gemm_wmma_bf16<<<grid, 256, lds, stream>>>(
            A, lda, W, ldw, (const float*)d_in[bias_idx],
            oB, oF, ldo, N, K, act);
    };

    // ---- stage 1: hashgrid + camera embed ----
    hash_input_kernel<<<(Bn + 255) / 256, 256, 0, stream>>>(
        y_in, x_in, cam, htab, in0, Bn);

    // ---- stage 2: pos MLP (SIREN x3) ----
    gemm(in0,  64,  posW0, 64,  7,  bufA, nullptr, 512, 512, 64,  ACT_SIREN);
    gemm(bufA, 512, posW1, 512, 9,  bufB, nullptr, 512, 512, 512, ACT_SIREN);
    gemm(bufB, 512, posW2, 512, 11, base, nullptr, 384, 384, 512, ACT_SIREN);

    // ---- stage 3: delta MLP (ReLU x4, linear) ----
    gemm(base, 384, dW[0], 384, 13, bufA, nullptr, 256, 256, 384, ACT_RELU);
    gemm(bufA, 256, dW[1], 256, 15, bufB, nullptr, 256, 256, 256, ACT_RELU);
    gemm(bufB, 256, dW[2], 256, 17, bufA, nullptr, 256, 256, 256, ACT_RELU);
    gemm(bufA, 256, dW[3], 256, 19, bufB, nullptr, 256, 256, 256, ACT_RELU);
    gemm(bufB, 256, dW[4], 256, 21, nullptr, delta_lin, 16, 9, 256, ACT_NONE);

    // ---- stage 4: dirs + PE + feature assembly (also writes delta_raw) ----
    assemble_kernel<<<(Bn + 255) / 256, 256, 0, stream>>>(
        delta_lin, ray_o, ray_d, base, feat0, feat1, feat2, dout, Bn);

    // ---- stage 5: color heads (SIREN, ReLU, ReLU, linear+sigmoid) ----
    u16* featp[3] = { feat0, feat1, feat2 };
    for (int c = 0; c < 3; ++c) {
        int bi = 22 + 8 * c;
        gemm(featp[c], 512, hW[c][0], 512, bi + 1, bufA, nullptr, 768, 768, 512, ACT_SIREN);
        gemm(bufA,     768, hW[c][1], 768, bi + 3, bufB, nullptr, 768, 768, 768, ACT_RELU);
        gemm(bufB,     768, hW[c][2], 768, bi + 5, bufA, nullptr, 768, 768, 768, ACT_RELU);
        gemm(bufA,     768, hW[c][3], 768, bi + 7, nullptr, hout[c], 16, 3, 768, ACT_SIGMOID);
    }

    // ---- stage 6: alpha head (ReLU, ReLU, linear+sigmoid); input == feat0 ----
    gemm(feat0, 512, aW0, 512, 47, bufA, nullptr, 256, 256, 512, ACT_RELU);
    gemm(bufA,  256, aW1, 256, 49, bufB, nullptr, 256, 256, 256, ACT_RELU);
    gemm(bufB,  256, aW2, 256, 51, nullptr, aout, 16, 1, 256, ACT_SIGMOID);

    // ---- stage 7: mix + pack outputs ----
    finalize_kernel<<<(Bn + 255) / 256, 256, 0, stream>>>(
        hout[0], hout[1], hout[2], aout, mixW, dout, Bn);
}
